// ConvGeodesic_1889785610349
// MI455X (gfx1250) — matrix operand order repaired and checked by reading, not compiled, not run on hardware
//
#include <hip/hip_runtime.h>
#include <hip/hip_bf16.h>
#include <stdint.h>

// ---------------- problem constants ----------------
#define BB 2
#define MM 30000
#define TT 8
#define RR 5
#define NN 64
#define OO 64
#define KK 2
#define RON (RR*OO*NN)          // 20480 elems of w_sum
#define NTILES ((BB*MM)/16)     // 3750 wmma m-tiles

typedef __attribute__((ext_vector_type(16))) __bf16 v16bf;
typedef __attribute__((ext_vector_type(8)))  __bf16 v8bf;
typedef __attribute__((ext_vector_type(8)))  float  v8f;

union ABfrag { v16bf v; v8bf h[2]; };

__device__ __forceinline__ unsigned short bf_bits(float f) {
  unsigned u = __builtin_bit_cast(unsigned, f);
  unsigned r = (u + 0x7FFFu + ((u >> 16) & 1u)) >> 16;   // RNE
  return (unsigned short)r;
}
__device__ __forceinline__ __bf16 to_bf(float f) {
  return __builtin_bit_cast(__bf16, bf_bits(f));
}

// lane-xor shuffle via ds_swizzle (wave32-native, group-of-32 mode)
#define SWZ_XOR_F(v, m) __builtin_bit_cast(float, \
    __builtin_amdgcn_ds_swizzle(__builtin_bit_cast(int, (v)), (((m) << 10) | 0x1f)))

// ---------------------------------------------------------------------------
// Kernel A: w_sum[r][o][n] = sum_{k,t} weights[k][t][r][o][n]  -> bf16
// Layout (R,O,N), n contiguous == exactly the WMMA B-fragment source layout.
// ---------------------------------------------------------------------------
__global__ void __launch_bounds__(256)
wsum_kernel(const float* __restrict__ weights, unsigned short* __restrict__ wsum)
{
  const int i = blockIdx.x * 256 + threadIdx.x;   // grid sized exactly: 80*256 == RON
  float s = 0.f;
#pragma unroll
  for (int kt = 0; kt < KK * TT; ++kt)
    s += weights[(size_t)kt * RON + i];
  wsum[i] = bf_bits(s);
}

// ---------------------------------------------------------------------------
// Kernel B: S2[b][r][m][o] = sum_n signal[b][m][n] * w_sum[r][o][n]   (bf16)
// One wave per 16-row m-tile. 5r * 2Ksteps * 4otiles = 40 v_wmma per tile.
// ---------------------------------------------------------------------------
__global__ void __launch_bounds__(256)
s2_wmma_kernel(const float* __restrict__ signal,
               const unsigned short* __restrict__ wsum,
               unsigned short* __restrict__ s2)
{
  __shared__ __align__(16) unsigned short lds_w[RON];   // 40 KB of 320 KB/WGP

  // cooperative 40 KB load, 16B per transaction
  {
    const uint4* src = (const uint4*)wsum;
    uint4*       dst = (uint4*)lds_w;
#pragma unroll
    for (int i = threadIdx.x; i < (RON * 2) / 16; i += 256) dst[i] = src[i];
  }
  __syncthreads();

  const int wave = threadIdx.x >> 5;
  const int lane = threadIdx.x & 31;
  const int tile = blockIdx.x * 8 + wave;
  if (tile >= NTILES) return;                 // wave-uniform: EXEC all-1 past here

  const int b    = tile / (MM / 16);
  const int m0   = (tile % (MM / 16)) * 16;
  const int row  = lane & 15;                 // A: matrix row / B,D: column-in-tile
  const int half = lane >> 4;

  // ---- A fragments (16x32 bf16, ISA layout: lo-lanes K 0-7 & 16-23) ----
  const float* srow = signal + ((size_t)b * MM + (m0 + row)) * NN;
  ABfrag afrag[2];
#pragma unroll
  for (int s = 0; s < 2; ++s) {
    const int n0 = s * 32 + half * 8;
    float4 p0 = *(const float4*)(srow + n0);
    float4 p1 = *(const float4*)(srow + n0 + 4);
    float4 p2 = *(const float4*)(srow + n0 + 16);
    float4 p3 = *(const float4*)(srow + n0 + 20);
    float tmp[16] = { p0.x,p0.y,p0.z,p0.w, p1.x,p1.y,p1.z,p1.w,
                      p2.x,p2.y,p2.z,p2.w, p3.x,p3.y,p3.z,p3.w };
#pragma unroll
    for (int e = 0; e < 16; ++e) afrag[s].v[e] = to_bf(tmp[e]);
  }

#pragma unroll
  for (int r = 0; r < RR; ++r) {
    v8f acc[4];
#pragma unroll
    for (int ot = 0; ot < 4; ++ot) acc[ot] = 0.0f;

#pragma unroll
    for (int s = 0; s < 2; ++s) {
      const int K0 = s * 32;
#pragma unroll
      for (int ot = 0; ot < 4; ++ot) {
        // B fragment (32x16 bf16): lane = column o, K 8h..8h+7 & 16+8h..16+8h+7
        const int oc = ot * 16 + row;
        const unsigned short* bp = lds_w + ((r * OO + oc) * NN + K0 + half * 8);
        ABfrag bfr;
        bfr.h[0] = *(const v8bf*)bp;        // K base..base+7
        bfr.h[1] = *(const v8bf*)(bp + 16); // K base+16..base+23
        acc[ot] = __builtin_amdgcn_wmma_f32_16x16x32_bf16(
            false, afrag[s].v, false, bfr.v, (short)0, acc[ot], false, false);
      }
    }

    // D layout: element i -> row m0 + i + half*8, column oc
    unsigned short* obase = s2 + ((size_t)b * RR + r) * (size_t)MM * OO;
#pragma unroll
    for (int ot = 0; ot < 4; ++ot) {
      const int oc = ot * 16 + row;
#pragma unroll
      for (int i = 0; i < 8; ++i)
        obase[(size_t)(m0 + i + half * 8) * OO + oc] = bf_bits(acc[ot][i]);
    }
  }
}

// ---------------------------------------------------------------------------
// Kernel C: one wave per (b,m).  conv[t][o] = sum_{r,c} w * S2[b,r,idx,o],
// relu, norm^2 wave-reduce, argmax over t (strict > == first max), +bias.
// Lane owns o = {2*lane, 2*lane+1}; each S2 row read = one coalesced b32/lane.
// ---------------------------------------------------------------------------
__global__ void __launch_bounds__(256)
pool_kernel(const float* __restrict__ bary,
            const unsigned short* __restrict__ s2,
            const float* __restrict__ bias,
            float* __restrict__ out)
{
  const int wave = threadIdx.x >> 5;
  const int lane = threadIdx.x & 31;
  const int bm   = blockIdx.x * 8 + wave;    // grid sized exactly: < B*M
  const int b    = bm / MM;
  const int m    = bm % MM;

  const int j  = lane < 15 ? lane : 14;      // lanes 0..14 stage the 15 (w,idx) pairs
  const int rj = j / 3, cj = j % 3;

  float best_n = -1.f, ba0 = 0.f, ba1 = 0.f;

  for (int t = 0; t < TT; ++t) {
    const float* bb = bary + ((size_t)bm * TT + t) * (RR * 8);
    float wv   = bb[rj * 8 + cj * 2];
    float fidx = bb[rj * 8 + cj * 2 + 1];

    float a0 = 0.f, a1 = 0.f;
#pragma unroll
    for (int jj = 0; jj < 15; ++jj) {
      const int r = jj / 3;                  // compile-time
      float w = __builtin_bit_cast(float,
                  __builtin_amdgcn_readlane(__builtin_bit_cast(int, wv), jj));
      int idx = (int)__builtin_bit_cast(float,
                  __builtin_amdgcn_readlane(__builtin_bit_cast(int, fidx), jj));
      const unsigned short* rp = s2 + (((size_t)b * RR + r) * MM + idx) * OO;
      unsigned u = *(const unsigned*)(rp + 2 * lane);
      a0 += w * __builtin_bit_cast(float, u << 16);
      a1 += w * __builtin_bit_cast(float, u & 0xFFFF0000u);
    }
    a0 = fmaxf(a0, 0.f);
    a1 = fmaxf(a1, 0.f);

    float n2 = a0 * a0 + a1 * a1;
    n2 += SWZ_XOR_F(n2, 16);
    n2 += SWZ_XOR_F(n2, 8);
    n2 += SWZ_XOR_F(n2, 4);
    n2 += SWZ_XOR_F(n2, 2);
    n2 += SWZ_XOR_F(n2, 1);

    if (n2 > best_n) { best_n = n2; ba0 = a0; ba1 = a1; }
  }

  float2 bs = *(const float2*)(bias + (size_t)m * OO + 2 * lane);
  float2 o;  o.x = ba0 + bs.x;  o.y = ba1 + bs.y;
  *(float2*)(out + (size_t)bm * OO + 2 * lane) = o;
}

// ---------------------------------------------------------------------------
extern "C" void kernel_launch(void* const* d_in, const int* in_sizes, int n_in,
                              void* d_out, int out_size, void* d_ws, size_t ws_size,
                              hipStream_t stream) {
  const float* signal  = (const float*)d_in[0];   // (B,M,N)      f32
  const float* bary    = (const float*)d_in[1];   // (B,M,T,R,8)  f32
  const float* weights = (const float*)d_in[2];   // (K,T,R,O,N)  f32
  const float* bias    = (const float*)d_in[3];   // (M,O)        f32
  float* out = (float*)d_out;                     // (B,M,O)      f32

  // workspace: [0, 40960) w_sum bf16 ; [65536, 65536+38.4MB) S2 bf16
  unsigned short* wsum = (unsigned short*)d_ws;
  unsigned short* s2   = (unsigned short*)((char*)d_ws + 65536);

  wsum_kernel<<<RON / 256, 256, 0, stream>>>(weights, wsum);
  s2_wmma_kernel<<<(NTILES + 7) / 8, 256, 0, stream>>>(signal, wsum, s2);
  pool_kernel<<<(BB * MM) / 8, 256, 0, stream>>>(bary, s2, bias, out);
}
